// MambaAE_39075612459687
// MI455X (gfx1250) — matrix operand order repaired and checked by reading, not compiled, not run on hardware
//
#include <hip/hip_runtime.h>
#include <hip/hip_bf16.h>
#include <math.h>

typedef __attribute__((ext_vector_type(16))) _Float16 v16h;
typedef __attribute__((ext_vector_type(8)))  float    v8f;

#define D_MODEL 128
#define D_INNER 256
#define D_STATE 16
#define DT_RANK 8
#define D_CONV  4
#define DBL_N   40
#define DBL_STRIDE 48
#define OUT_N (2*128*128*128)

// packed-weight geometry: P[((ntile*ksteps + kstep)*32 + lane)*16 + e]
#define IN_NTILES  32
#define IN_KSTEPS  4    // K=128
#define XP_NTILES  3    // N=40 padded to 48
#define XP_KSTEPS  8    // K=256
#define OW_NTILES  8    // N=128
#define OW_KSTEPS  8    // K=256
#define IN_PACK_PER_DIR (IN_NTILES*IN_KSTEPS*32*16)   // 65536 halves
#define XP_PACK_PER_DIR (XP_NTILES*XP_KSTEPS*32*16)   // 12288 halves
#define OW_PACK_PER_DIR (OW_NTILES*OW_KSTEPS*32*16)   // 32768 halves

__device__ __forceinline__ float silu_f(float v){ return v / (1.0f + expf(-v)); }
__device__ __forceinline__ float softplus_f(float v){ return v > 20.0f ? v : log1pf(expf(v)); }

// direction mapping: scan token (s,t) -> spatial (r,c) within the 8x8 chunk.
__device__ __forceinline__ void map_rc(int d, int s, int t, int& r, int& c){
  if (d == 0){ r = s;     c = t;     }
  else if (d == 1){ r = s;     c = 7 - t; }
  else if (d == 2){ r = t;     c = s;     }
  else            { r = 7 - t; c = s;     }
}

// ---- WMMA fragment helpers (ISA 7.12.2 layouts, wave32) -------------------

// A fragment: 16x32 f16 tile of src[m][k] (row-major LDS, stride in halves).
__device__ __forceinline__ v16h load_A_lds(const _Float16* src, int stride,
                                           int mtile, int kbase, int lane){
  int m   = mtile * 16 + (lane & 15);
  int off = (lane & 16) ? 8 : 0;
  const _Float16* p = src + m * stride + kbase + off;
  union { uint4 q[2]; v16h v; } u;
  u.q[0] = *(const uint4*)(p);
  u.q[1] = *(const uint4*)(p + 16);
  return u.v;
}

// B fragment from pre-packed lane-layout weights: one 32B contiguous load.
__device__ __forceinline__ v16h load_B_packed(const _Float16* P, int ksteps,
                                              int ntile, int kstep, int lane){
  const _Float16* p = P + ((size_t)(ntile * ksteps + kstep) * 32 + lane) * 16;
  union { uint4 q[2]; v16h v; } u;
  u.q[0] = ((const uint4*)p)[0];
  u.q[1] = ((const uint4*)p)[1];
  return u.v;
}

// ---- kernels ---------------------------------------------------------------

// Pack f32 weights W (nmax x Krow, per-dir stride dir_stride) into WMMA
// B-fragment lane layout, zero-padding n >= nmax. One thread per f16 element.
__global__ void pack_b_kernel(const float* __restrict__ W, _Float16* __restrict__ P,
                              int Krow, int nmax, int ntiles, int dir_stride){
  int ksteps = Krow >> 5;
  int idx = blockIdx.x * 256 + threadIdx.x;
  int total = 4 * ntiles * ksteps * 32 * 16;
  if (idx >= total) return;
  int e     = idx & 15;
  int lane  = (idx >> 4) & 31;
  int rest  = idx >> 9;             // (dir*ntiles + ntile)*ksteps + kstep
  int kstep = rest % ksteps;
  int nt_d  = rest / ksteps;
  int ntile = nt_d % ntiles;
  int dir   = nt_d / ntiles;
  int n = ntile * 16 + (lane & 15);
  int k = kstep * 32 + ((lane & 16) ? 8 : 0) + (e & 7) + ((e & 8) ? 16 : 0);
  float v = (n < nmax) ? W[(size_t)dir * dir_stride + (size_t)n * Krow + k] : 0.0f;
  P[idx] = (_Float16)v;
}

__global__ void stats_kernel(const float* __restrict__ x, float* __restrict__ stats){
  __shared__ float rs[256], rq[256];
  int bc = blockIdx.x;                       // (b*128 + c)
  const float* p = x + (size_t)bc * 16384;
  float s = 0.f, q = 0.f;
  for (int i = threadIdx.x; i < 16384; i += 256){ float v = p[i]; s += v; q += v*v; }
  rs[threadIdx.x] = s; rq[threadIdx.x] = q;
  __syncthreads();
  for (int o = 128; o > 0; o >>= 1){
    if ((int)threadIdx.x < o){ rs[threadIdx.x] += rs[threadIdx.x+o]; rq[threadIdx.x] += rq[threadIdx.x+o]; }
    __syncthreads();
  }
  if (threadIdx.x == 0){
    float mean = rs[0] * (1.0f/16384.0f);
    float var  = rq[0] * (1.0f/16384.0f) - mean*mean;
    stats[bc*2+0] = mean;
    stats[bc*2+1] = rsqrtf(var + 1e-5f);
  }
}

__global__ void __launch_bounds__(256)
mamba_mega(const float* __restrict__ x, const float* __restrict__ gamma,
           const float* __restrict__ beta, const float* __restrict__ stats,
           const _Float16* __restrict__ pw_in, const float* __restrict__ conv_w,
           const float* __restrict__ conv_b, const _Float16* __restrict__ pw_xp,
           const float* __restrict__ dt_w, const float* __restrict__ dt_b,
           const float* __restrict__ A_log, const float* __restrict__ Dskip,
           const _Float16* __restrict__ pw_ow, const float* __restrict__ alpha,
           float* __restrict__ out)
{
  extern __shared__ char smem[];
  _Float16* sA   = (_Float16*)smem;                            // 64x128 tokens (scan order)
  _Float16* sXZ  = (_Float16*)(smem + 16384);                  // 64x512: xh(->xc) | z
  float*    sDBL = (float*)  (smem + 16384 + 65536);           // 64x48
  _Float16* sY   = (_Float16*)(smem + 16384 + 65536 + 12288);  // 64x256 gated y

  const int tid  = threadIdx.x;
  const int lane = tid & 31;
  const int wave = tid >> 5;
  const int dir   = blockIdx.x & 3;
  const int chunk = blockIdx.x >> 2;
  const int b  = chunk >> 8;
  const int hi = (chunk >> 4) & 15;
  const int wi = chunk & 15;

  const _Float16* pwin_d = pw_in + (size_t)dir * IN_PACK_PER_DIR;
  const _Float16* pwxp_d = pw_xp + (size_t)dir * XP_PACK_PER_DIR;
  const _Float16* pwow_d = pw_ow + (size_t)dir * OW_PACK_PER_DIR;
  const float* cwd   = conv_w + dir * D_INNER * D_CONV;
  const float* cbd   = conv_b + dir * D_INNER;
  const float* dtwd  = dt_w   + dir * D_INNER * DT_RANK;
  const float* dtbd  = dt_b   + dir * D_INNER;
  const float* alogd = A_log  + dir * D_INNER * D_STATE;
  const float* dskd  = Dskip  + dir * D_INNER;
  const float  alpha0 = alpha[0];

  // warm L2/L0 with this direction's packed weights (global_prefetch_b8)
  __builtin_prefetch((const char*)pwin_d + tid * 512, 0, 0);
  __builtin_prefetch((const char*)pwow_d + tid * 256, 0, 0);

  // ---- phase 0: normalized, SiLU'd tokens in scan order (64 x 128 f16) ----
  for (int idx = tid; idx < 64*128; idx += 256){
    int m = idx >> 7, ch = idx & 127;
    int s = m >> 3, t = m & 7, r, c;
    map_rc(dir, s, t, r, c);
    size_t gi = (((size_t)b*D_MODEL + ch)*128 + hi*8 + r)*128 + wi*8 + c;
    float mean = stats[(b*D_MODEL + ch)*2 + 0];
    float rstd = stats[(b*D_MODEL + ch)*2 + 1];
    float v = (x[gi] - mean) * rstd;
    v = gamma[ch]*v + beta[ch];
    sA[idx] = (_Float16)silu_f(v);
  }
  __syncthreads();

  // ---- GEMM1: xz(64x512) = tok(64x128) * in_w^T, each wave owns 4 N-tiles --
  for (int jj = 0; jj < 4; jj++){
    int ntile = wave*4 + jj;
    v8f acc[4];
#pragma unroll
    for (int mt = 0; mt < 4; mt++)
#pragma unroll
      for (int i = 0; i < 8; i++) acc[mt][i] = 0.0f;
#pragma unroll
    for (int k4 = 0; k4 < IN_KSTEPS; k4++){
      v16h bf = load_B_packed(pwin_d, IN_KSTEPS, ntile, k4, lane);
#pragma unroll
      for (int mt = 0; mt < 4; mt++){
        v16h af = load_A_lds(sA, D_MODEL, mt, k4*32, lane);
        acc[mt] = __builtin_amdgcn_wmma_f32_16x16x32_f16(
            false, af, false, bf, (short)0, acc[mt], false, false);
      }
    }
    int n  = ntile*16 + (lane & 15);
    int mo = (lane & 16) ? 8 : 0;
#pragma unroll
    for (int mt = 0; mt < 4; mt++)
#pragma unroll
      for (int i = 0; i < 8; i++)
        sXZ[(mt*16 + mo + i)*512 + n] = (_Float16)acc[mt][i];
  }
  __syncthreads();

  // ---- causal depthwise conv (K=4) + SiLU, in place over xh half ----------
  for (int p = tid; p < 8*D_INNER; p += 256){
    int s = p >> 8, ch = p & 255;
    float w0 = cwd[ch*4+0], w1 = cwd[ch*4+1], w2 = cwd[ch*4+2], w3 = cwd[ch*4+3];
    float bb = cbd[ch];
    float xh[8];
#pragma unroll
    for (int t = 0; t < 8; t++) xh[t] = (float)sXZ[(s*8+t)*512 + ch];
#pragma unroll
    for (int t = 0; t < 8; t++){
      float a = bb + xh[t]*w3;
      if (t >= 1) a += xh[t-1]*w2;
      if (t >= 2) a += xh[t-2]*w1;
      if (t >= 3) a += xh[t-3]*w0;
      sXZ[(s*8+t)*512 + ch] = (_Float16)silu_f(a);
    }
  }
  __syncthreads();

  // ---- GEMM2: dbl(64x40) = xc(64x256) * xproj_w^T (12 tiles over waves 0..5)
  if (wave < 6){
    for (int q = 0; q < 2; q++){
      int tt = wave*2 + q;
      int mt = tt / 3, nt = tt % 3;
      v8f acc;
#pragma unroll
      for (int i = 0; i < 8; i++) acc[i] = 0.0f;
#pragma unroll
      for (int k8 = 0; k8 < XP_KSTEPS; k8++){
        v16h bf = load_B_packed(pwxp_d, XP_KSTEPS, nt, k8, lane);
        v16h af = load_A_lds(sXZ, 512, mt, k8*32, lane);
        acc = __builtin_amdgcn_wmma_f32_16x16x32_f16(
            false, af, false, bf, (short)0, acc, false, false);
      }
      int n  = nt*16 + (lane & 15);
      int mo = (lane & 16) ? 8 : 0;
      if (n < DBL_N){
#pragma unroll
        for (int i = 0; i < 8; i++) sDBL[(mt*16 + mo + i)*DBL_STRIDE + n] = acc[i];
      }
    }
  }
  __syncthreads();

  // ---- fused dt_proj + softplus + selective scan + skip + gating ----------
  for (int p = tid; p < 8*D_INNER; p += 256){
    int s = p >> 8, d = p & 255;
    float dtw_[DT_RANK];
#pragma unroll
    for (int r = 0; r < DT_RANK; r++) dtw_[r] = dtwd[d*DT_RANK + r];
    float dtb_ = dtbd[d];
    float Av[D_STATE];
#pragma unroll
    for (int st = 0; st < D_STATE; st++) Av[st] = -expf(alogd[d*D_STATE + st]);
    float Dp = dskd[d];
    float h[D_STATE];
#pragma unroll
    for (int st = 0; st < D_STATE; st++) h[st] = 0.0f;
    for (int t = 0; t < 8; t++){
      int m = s*8 + t;
      const float* dr = sDBL + m*DBL_STRIDE;
      float dtv = dtb_;
#pragma unroll
      for (int r = 0; r < DT_RANK; r++) dtv += dr[r]*dtw_[r];
      dtv = softplus_f(dtv);
      float xt = (float)sXZ[m*512 + d];
      float dx = dtv * xt;
      float y = 0.0f;
#pragma unroll
      for (int st = 0; st < D_STATE; st++){
        float dA = expf(dtv * Av[st]);
        h[st] = dA*h[st] + dx*dr[8+st];
        y += h[st]*dr[24+st];
      }
      float z = (float)sXZ[m*512 + D_INNER + d];
      sY[m*D_INNER + d] = (_Float16)((y + Dp*xt) * silu_f(z));
    }
  }
  __syncthreads();

  // ---- GEMM3: out(64x128) = yg(64x256) * out_w^T; scaled atomic into out ---
  {
    int mt  = wave >> 1;
    int nt0 = (wave & 1)*4;
    v8f acc[4];
#pragma unroll
    for (int j = 0; j < 4; j++)
#pragma unroll
      for (int i = 0; i < 8; i++) acc[j][i] = 0.0f;
#pragma unroll
    for (int k8 = 0; k8 < OW_KSTEPS; k8++){
      v16h af = load_A_lds(sY, D_INNER, mt, k8*32, lane);
#pragma unroll
      for (int j = 0; j < 4; j++){
        v16h bf = load_B_packed(pwow_d, OW_KSTEPS, nt0+j, k8, lane);
        acc[j] = __builtin_amdgcn_wmma_f32_16x16x32_f16(
            false, af, false, bf, (short)0, acc[j], false, false);
      }
    }
    int mo = (lane & 16) ? 8 : 0;
#pragma unroll
    for (int j = 0; j < 4; j++){
      int ch = (nt0+j)*16 + (lane & 15);
#pragma unroll
      for (int i = 0; i < 8; i++){
        int m = mt*16 + mo + i;
        int s = m >> 3, t = m & 7, r, c;
        map_rc(dir, s, t, r, c);
        size_t gi = (((size_t)b*D_MODEL + ch)*128 + hi*8 + r)*128 + wi*8 + c;
        atomicAdd(&out[gi], alpha0 * acc[j][i]);
      }
    }
  }
}

extern "C" void kernel_launch(void* const* d_in, const int* in_sizes, int n_in,
                              void* d_out, int out_size, void* d_ws, size_t ws_size,
                              hipStream_t stream) {
  const float* x      = (const float*)d_in[0];
  const float* gamma  = (const float*)d_in[1];
  const float* beta   = (const float*)d_in[2];
  const float* alpha  = (const float*)d_in[3];
  const float* in_w   = (const float*)d_in[4];
  const float* conv_w = (const float*)d_in[5];
  const float* conv_b = (const float*)d_in[6];
  const float* xpw    = (const float*)d_in[7];
  const float* dtw    = (const float*)d_in[8];
  const float* dtb    = (const float*)d_in[9];
  const float* alog   = (const float*)d_in[10];
  const float* dsk    = (const float*)d_in[11];
  const float* outw   = (const float*)d_in[12];
  float* out = (float*)d_out;

  // ws layout: stats | packed in_proj | packed x_proj | packed out_proj
  float*    stats = (float*)d_ws;
  _Float16* pw_in = (_Float16*)((char*)d_ws + 4096);
  _Float16* pw_xp = pw_in + 4 * IN_PACK_PER_DIR;
  _Float16* pw_ow = pw_xp + 4 * XP_PACK_PER_DIR;

  stats_kernel<<<256, 256, 0, stream>>>(x, stats);
  pack_b_kernel<<<(4*IN_PACK_PER_DIR + 255)/256, 256, 0, stream>>>(
      in_w, pw_in, D_MODEL, 2*D_INNER, IN_NTILES, 2*D_INNER*D_MODEL);
  pack_b_kernel<<<(4*XP_PACK_PER_DIR + 255)/256, 256, 0, stream>>>(
      xpw, pw_xp, D_INNER, DBL_N, XP_NTILES, DBL_N*D_INNER);
  pack_b_kernel<<<(4*OW_PACK_PER_DIR + 255)/256, 256, 0, stream>>>(
      outw, pw_ow, D_INNER, D_MODEL, OW_NTILES, D_MODEL*D_INNER);

  // residual: out = x, then mega kernel accumulates alpha * vss into it
  hipMemcpyAsync(out, x, (size_t)OUT_N * sizeof(float),
                 hipMemcpyDeviceToDevice, stream);

  const size_t smem = 16384 + 65536 + 12288 + 32768; // 124 KB
  mamba_mega<<<2048, 256, smem, stream>>>(x, gamma, beta, stats,
                                          pw_in, conv_w, conv_b, pw_xp, dtw, dtb,
                                          alog, dsk, pw_ow, alpha, out);
}
